// SimpleLSTM_47382079209494
// MI455X (gfx1250) — compile-verified, matrix operand order checked by reading, not run on hardware
//
#include <hip/hip_runtime.h>
#include <hip/hip_bf16.h>

typedef __attribute__((ext_vector_type(16))) _Float16 v16h;
typedef __attribute__((ext_vector_type(8)))  _Float16 v8h;
typedef __attribute__((ext_vector_type(8)))  float    v8f;
typedef __attribute__((ext_vector_type(4)))  float    v4f;

#define B_   8192
#define T_   512
#define I_   12
#define H_   64
#define G_   256   // 4*H
#define C_   18
#define BT   16    // batch rows per workgroup
#define GROW 260   // padded gate row (floats) to break bank conflicts
#define HROW 72    // padded h row (halves)

#define LOG2E  1.4426950408889634f
#define LOG2E2 2.8853900817779268f

// Single-TRANS-op activations: v_exp_f32 (= exp2) + v_rcp_f32.
// exp2 saturates to 0/inf, rcp(inf)=0 -> correct asymptotes for gates.
__device__ __forceinline__ float fast_sigmoid(float x) {
    float e = __builtin_amdgcn_exp2f(-LOG2E * x);
    return __builtin_amdgcn_rcpf(1.0f + e);
}
__device__ __forceinline__ float fast_tanh(float x) {
    float e = __builtin_amdgcn_exp2f(-LOG2E2 * x);
    return 2.0f * __builtin_amdgcn_rcpf(1.0f + e) - 1.0f;
}

__global__ __launch_bounds__(256)
void lstm_fused_kernel(const float* __restrict__ x,
                       const float* __restrict__ W_ih,
                       const float* __restrict__ W_hh,
                       const float* __restrict__ b_ih,
                       const float* __restrict__ b_hh,
                       const float* __restrict__ fcW,
                       const float* __restrict__ fcb,
                       float* __restrict__ out)
{
    __shared__ float    gates[BT][GROW];   // 16.6 KB
    __shared__ _Float16 hbuf[BT][HROW];    // 2.3 KB

    const int tid   = threadIdx.x;
    const int wave  = tid >> 5;            // 0..7 : gate-column slice of 32
    const int lane  = tid & 31;
    const int lnN   = lane & 15;           // N (or M) within a 16-wide tile
    const int hiK   = lane >> 4;           // 0: lanes 0-15, 1: lanes 16-31
    const int kOff  = hiK * 8;             // K sub-offset per ISA 16-bit layout
    const int b0    = blockIdx.x * BT;
    const int nBase = wave * 32;

    // ---------------- persistent B fragments (live in VGPRs all 512 steps) ----
    // B(k,n): W_hh part (k<64 over 2 chunks) and x-extension chunk
    // (rows 0-11 = W_ih^T, row 12 = b_ih+b_hh, rest 0).
    v16h Bh[2][2];
    v16h Bx[2];
    #pragma unroll
    for (int nt = 0; nt < 2; ++nt) {
        const int n = nBase + nt * 16 + lnN;       // gate column 0..255
        const float* wr = W_hh + n * H_;           // W_hh[n][k], k contiguous
        #pragma unroll
        for (int kc = 0; kc < 2; ++kc) {
            const int k0 = kc * 32 + kOff;
            v16h f;
            #pragma unroll
            for (int e = 0; e < 16; ++e) {
                const int k = k0 + e + ((e >= 8) ? 8 : 0);   // ISA 16-bit frag order
                f[e] = (_Float16)wr[k];
            }
            Bh[nt][kc] = f;
        }
        const float bias = b_ih[n] + b_hh[n];
        const float* wi = W_ih + n * I_;
        v16h fx;
        #pragma unroll
        for (int e = 0; e < 16; ++e) {
            const int k = kOff + e + ((e >= 8) ? 8 : 0);
            float v = 0.0f;
            if (k < I_)        v = wi[k];
            else if (k == I_)  v = bias;               // multiplied by A's 1.0 lane
            fx[e] = (_Float16)v;
        }
        Bx[nt] = fx;
    }

    // zero-init h state
    for (int i = tid; i < BT * HROW; i += 256)
        ((_Float16*)hbuf)[i] = (_Float16)0.0f;

    // cell state lives in registers of the pointwise owner thread
    float cst[4] = {0.f, 0.f, 0.f, 0.f};
    const int pcol  = tid & 63;            // h column this thread updates
    const int prow0 = (tid >> 6) * 4;      // 4 batch rows per thread

    const float* xrow = x + (long)(b0 + lnN) * T_ * I_;

    __syncthreads();

    for (int t = 0; t < T_; ++t) {
        // ---- A fragments: h (2 K-chunks) from LDS, ISA 16-bit A layout -------
        v16h Ah[2];
        #pragma unroll
        for (int kc = 0; kc < 2; ++kc) {
            const int kb = kc * 32 + kOff;
            v8h lo = *(const v8h*)&hbuf[lnN][kb];        // ds_load_b128
            v8h hi = *(const v8h*)&hbuf[lnN][kb + 16];   // ds_load_b128
            v16h f;
            #pragma unroll
            for (int e = 0; e < 8; ++e) { f[e] = lo[e]; f[8 + e] = hi[e]; }
            Ah[kc] = f;
        }
        // ---- A fragment: [x(12), 1.0, 0...] straight from global -------------
        // Branchless: both halves load unconditionally (always in-bounds),
        // hi half swaps in {1.0(bias), 0,0,0} via cndmask.
        const float* xp = xrow + t * I_;
        v4f p0  = *(const v4f*)(xp + kOff);              // x[0..3] or x[8..11]
        v4f p1l = *(const v4f*)(xp + 4);                 // x[4..7]
        v4f p1;
        p1[0] = hiK ? 1.0f : p1l[0];                     // k=12 bias multiplier
        p1[1] = hiK ? 0.0f : p1l[1];
        p1[2] = hiK ? 0.0f : p1l[2];
        p1[3] = hiK ? 0.0f : p1l[3];
        v16h Ax;
        #pragma unroll
        for (int e = 0; e < 4; ++e) {
            Ax[e]      = (_Float16)p0[e];
            Ax[4 + e]  = (_Float16)p1[e];
            Ax[8 + e]  = (_Float16)0.0f;                 // K>=16 half of chunk: zero
            Ax[12 + e] = (_Float16)0.0f;
        }

        // ---- gates tile = [x,1]*Wx + h*Whh : 6 WMMAs per wave ---------------
        #pragma unroll
        for (int nt = 0; nt < 2; ++nt) {
            v8f acc = {};
            acc = __builtin_amdgcn_wmma_f32_16x16x32_f16(false, Ax,    false, Bx[nt],
                                                         (short)0, acc, false, false);
            acc = __builtin_amdgcn_wmma_f32_16x16x32_f16(false, Ah[0], false, Bh[nt][0],
                                                         (short)0, acc, false, false);
            acc = __builtin_amdgcn_wmma_f32_16x16x32_f16(false, Ah[1], false, Bh[nt][1],
                                                         (short)0, acc, false, false);
            const int ncol = nBase + nt * 16 + lnN;
            const int mrow = hiK * 8;                    // C/D layout: M=r / M=8+r
            #pragma unroll
            for (int r = 0; r < 8; ++r) gates[mrow + r][ncol] = acc[r];
        }
        __syncthreads();

        // ---- pointwise LSTM cell update (c in registers, h -> f16 LDS) -------
        #pragma unroll
        for (int r = 0; r < 4; ++r) {
            const int row = prow0 + r;
            const float ig = fast_sigmoid(gates[row][pcol]);
            const float fg = fast_sigmoid(gates[row][64  + pcol]);
            const float gg = fast_tanh   (gates[row][128 + pcol]);
            const float og = fast_sigmoid(gates[row][192 + pcol]);
            cst[r] = fg * cst[r] + ig * gg;
            hbuf[row][pcol] = (_Float16)(og * fast_tanh(cst[r]));
        }
        __syncthreads();
    }

    // ---------------- classifier head on h_T ---------------------------------
    for (int idx = tid; idx < BT * C_; idx += 256) {
        const int row = idx / C_;
        const int cls = idx % C_;
        const float* w = fcW + cls * H_;
        float s = fcb[cls];
        #pragma unroll 8
        for (int k = 0; k < H_; ++k) s += (float)hbuf[row][k] * w[k];
        out[(long)(b0 + row) * C_ + cls] = s;
    }
}

extern "C" void kernel_launch(void* const* d_in, const int* in_sizes, int n_in,
                              void* d_out, int out_size, void* d_ws, size_t ws_size,
                              hipStream_t stream) {
    const float* x    = (const float*)d_in[0];
    const float* W_ih = (const float*)d_in[1];
    const float* W_hh = (const float*)d_in[2];
    const float* b_ih = (const float*)d_in[3];
    const float* b_hh = (const float*)d_in[4];
    const float* fcW  = (const float*)d_in[5];
    const float* fcb  = (const float*)d_in[6];
    float* out = (float*)d_out;

    lstm_fused_kernel<<<B_ / BT, 256, 0, stream>>>(x, W_ih, W_hh, b_ih, b_hh,
                                                   fcW, fcb, out);
}
